// Optimized_MHSA_Block_21328807592209
// MI455X (gfx1250) — matrix operand order repaired and checked by reading, not compiled, not run on hardware
//
#include <hip/hip_runtime.h>
#include <hip/hip_bf16.h>

// ---------------------------------------------------------------------------
// Types for CDNA5 WMMA (wave32, V_WMMA_F32_16X16X32_BF16)
// ---------------------------------------------------------------------------
typedef __bf16 bf16_t;
typedef __attribute__((ext_vector_type(16))) __bf16 bf16x16;
typedef __attribute__((ext_vector_type(8)))  __bf16 bf16x8;
typedef __attribute__((ext_vector_type(8)))  float  f32x8;
typedef unsigned int u32;
typedef __attribute__((ext_vector_type(4))) u32 u32x4;
typedef __attribute__((ext_vector_type(8))) int  i32x8;
typedef __attribute__((ext_vector_type(4))) int  i32x4;

union BF16x16U { bf16x16 v; bf16x8 h[2]; };

__device__ __forceinline__ int lane_id() { return threadIdx.x & 31; }

// A-matrix 16x32 bf16 fragment from a row-major tile (ld in elements).
// Lanes 0-15 rows M=0..15; K = [8g..8g+7] and [16+8g..23+8g], g = lane>=16.
__device__ __forceinline__ bf16x16 load_frag_a(const bf16_t* base, int ld) {
  const int l = lane_id();
  const int m  = l & 15;
  const int ko = (l >> 4) * 8;
  const bf16_t* p = base + m * ld + ko;
  BF16x16U u;
  u.h[0] = *(const bf16x8*)(p);
  u.h[1] = *(const bf16x8*)(p + 16);
  return u.v;
}

// B-matrix 32x16 bf16 fragment from a TRANSPOSED tile bt[n][k] (ld elements).
// Lane holds col N = l&15, K = [8g..8g+7] and [16+8g..23+8g], g = lane>=16.
__device__ __forceinline__ bf16x16 load_frag_b_t(const bf16_t* bt, int ld) {
  const int l = lane_id();
  const int n  = l & 15;
  const int kb = (l >> 4) * 8;
  const bf16_t* p = bt + n * ld + kb;
  BF16x16U u;
  u.h[0] = *(const bf16x8*)(p);
  u.h[1] = *(const bf16x8*)(p + 16);
  return u.v;
}

__device__ __forceinline__ f32x8 wmma_bf16(bf16x16 a, bf16x16 b, f32x8 c) {
  return __builtin_amdgcn_wmma_f32_16x16x32_bf16(
      /*neg_a=*/false, a, /*neg_b=*/false, b,
      /*c_mod=*/(short)0, c, /*reuse_a=*/false, /*reuse_b=*/false);
}

// ---------------------------------------------------------------------------
// Tensor Data Mover (forced on: a missing builtin must fail the compile so
// the fallback can never silently mask it). Descriptor per
// CDNA5 ISA, D# groups 0/1.
// ---------------------------------------------------------------------------
#define USE_TDM 1

#if USE_TDM
// D# group 1, built once per kernel (loop-invariant).
//   tile_d0: contiguous elements per row; tile_d1: rows;
//   stride_el: source row stride (elements);
//   pad code iv: pad after 2^(iv+1) DWORDs; pad code am: insert am+1 DWORDs.
__device__ __forceinline__ i32x8 tdm_make_g1(u32 tile_d0, u32 tile_d1,
                                             u32 stride_el, u32 pad_en,
                                             u32 pad_iv, u32 pad_am) {
  i32x8 g1;
  g1[0] = (int)((1u << 16)                      // data_size = 1 (2 bytes)
                | (pad_en << 20)                // pad_enable
                | (pad_iv << 22) | (pad_am << 25));
  g1[1] = (int)(tile_d0 << 16);                 // tensor_dim0 lo16 @ bits 63:48
  g1[2] = (int)((tile_d0 >> 16) | (tile_d1 << 16));  // td0 hi | td1 lo
  g1[3] = (int)((tile_d1 >> 16) | (tile_d0 << 16));  // td1 hi | tile_dim0
  g1[4] = (int)(tile_d1 & 0xFFFFu);             // tile_dim1 (tile_dim2 = 0)
  g1[5] = (int)stride_el;                       // tensor_dim0_stride[31:0]
  g1[6] = 0;
  g1[7] = 0;
  return g1;
}

__device__ __forceinline__ u32x4 tdm_make_g0(const void* lds_dst,
                                             const void* gaddr) {
  const u32 lds_off =
      (u32)(unsigned long long)(__attribute__((address_space(3))) char*)lds_dst;
  const unsigned long long ga = (unsigned long long)gaddr;
  u32x4 g0;
  g0[0] = 1u;                                   // count=1 (valid), user mode
  g0[1] = lds_off;                              // lds_addr
  g0[2] = (u32)ga;                              // global_addr[31:0]
  g0[3] = ((u32)(ga >> 32) & 0x01FFFFFFu) | (2u << 30);  // addr[56:32]|type=2
  return g0;
}

__device__ __forceinline__ void tdm_load(void* lds_dst, const void* gsrc,
                                         i32x8 g1) {
  const u32x4 g0 = tdm_make_g0(lds_dst, gsrc);
  const i32x4 z4 = {0, 0, 0, 0};
#if __clang_major__ >= 23
  const i32x8 z8 = {0, 0, 0, 0, 0, 0, 0, 0};
  __builtin_amdgcn_tensor_load_to_lds(g0, g1, z4, z4, z8, 0);
#else
  __builtin_amdgcn_tensor_load_to_lds(g0, g1, z4, z4, 0);
#endif
}

__device__ __forceinline__ void tdm_store(const void* lds_src, void* gdst,
                                          i32x8 g1) {
  const u32x4 g0 = tdm_make_g0(lds_src, gdst);
  const i32x4 z4 = {0, 0, 0, 0};
#if __clang_major__ >= 23
  const i32x8 z8 = {0, 0, 0, 0, 0, 0, 0, 0};
  __builtin_amdgcn_tensor_store_from_lds(g0, g1, z4, z4, z8, 0);
#else
  __builtin_amdgcn_tensor_store_from_lds(g0, g1, z4, z4, 0);
#endif
}
#endif  // USE_TDM

// ---------------------------------------------------------------------------
// Transpose-convert for weights: src [R][C] fp32 -> dst [C][R] bf16.
// ---------------------------------------------------------------------------
template <int R, int C>
__global__ void cvt_t_kernel(const float* __restrict__ s,
                             bf16_t* __restrict__ d) {
  const size_t base = (size_t)blockIdx.y * (R * C);
  const int i = blockIdx.x * 256 + threadIdx.x;
  if (i < R * C) {
    const int r = i / C, c = i % C;
    d[base + (size_t)c * R + r] = (bf16_t)s[base + i];
  }
}

// ---------------------------------------------------------------------------
// LayerNorm over D=1024, one block per row, bf16 output
// ---------------------------------------------------------------------------
__global__ __launch_bounds__(256) void ln_kernel(const float* __restrict__ x,
                                                 const float* __restrict__ g,
                                                 const float* __restrict__ bt,
                                                 bf16_t* __restrict__ o) {
  __shared__ float rs[256];
  __shared__ float rq[256];
  const int t = threadIdx.x;
  const size_t row = blockIdx.x;
  const float* xr = x + row * 1024;
  float v[4], s = 0.f, q = 0.f;
#pragma unroll
  for (int i = 0; i < 4; ++i) {
    v[i] = xr[t + i * 256];
    s += v[i];
    q += v[i] * v[i];
  }
  rs[t] = s; rq[t] = q;
  __syncthreads();
  for (int off = 128; off > 0; off >>= 1) {
    if (t < off) { rs[t] += rs[t + off]; rq[t] += rq[t + off]; }
    __syncthreads();
  }
  const float mean = rs[0] * (1.f / 1024.f);
  const float var  = rq[0] * (1.f / 1024.f) - mean * mean;
  const float rstd = rsqrtf(var + 1e-5f);
#pragma unroll
  for (int i = 0; i < 4; ++i) {
    const int c = t + i * 256;
    o[row * 1024 + c] = (bf16_t)((v[i] - mean) * rstd * g[c] + bt[c]);
  }
}

// ---------------------------------------------------------------------------
// bf16 GEMM: C = A[M,K] @ Bt[N,K]^T. 128x128 block tile, 8 waves (4x2),
// wave tile 32x64, BK=32. Double-buffered LDS pipeline: wave 0 issues the
// next tile's TDM loads, then s_wait_tensorcnt 2 (TDM retires in order per
// wave) guarantees the current tile is resident -> DMA overlaps WMMA.
// MODE 1: bf16 head layout [B,H,S,hd] + transposed copy [B,H,hd,S]
// MODE 2: fp32 = acc + bias + res
// MODE 3: bf16 = gelu(acc + bias)
// ---------------------------------------------------------------------------
template <int MODE>
__global__ __launch_bounds__(256) void gemm_kernel(
    const bf16_t* __restrict__ A, const bf16_t* __restrict__ Bt,
    const float* __restrict__ bias, const float* __restrict__ res,
    void* __restrict__ outp, void* __restrict__ outp2, int M, int N, int K) {
  constexpr int LDT = 40;  // 32 data + 8 pad elements (80B rows, 16B aligned)
  __shared__ __align__(16) bf16_t As[2][128 * LDT];
  __shared__ __align__(16) bf16_t Bs[2][128 * LDT];

  const int t = threadIdx.x;
  const int w = t >> 5, l = t & 31;
  const int wm = w >> 1, wn = w & 1;
  const int l15 = l & 15, lg = l >> 4;
  const int m0 = blockIdx.y * 128, n0 = blockIdx.x * 128;
  const int NT = K >> 5;

  const f32x8 zero = {0.f, 0.f, 0.f, 0.f, 0.f, 0.f, 0.f, 0.f};
  f32x8 acc[2][4];
#pragma unroll
  for (int i = 0; i < 2; ++i)
#pragma unroll
    for (int j = 0; j < 4; ++j) acc[i][j] = zero;

#if USE_TDM
  // 128 rows x 32 el, row stride K; pad every 16 DWORDs by 4 DWORDs -> LDT=40
  const i32x8 g1t = tdm_make_g1(32, 128, (u32)K, 1, 3, 3);
  if (w == 0) {  // preload tile 0
    tdm_load(As[0], A + (size_t)m0 * K, g1t);
    tdm_load(Bs[0], Bt + (size_t)n0 * K, g1t);
  }
#endif

  for (int it = 0; it < NT; ++it) {
    const int buf = it & 1;
#if USE_TDM
    if (w == 0) {
      if (it + 1 < NT) {
        const int kn = (it + 1) << 5;
        tdm_load(As[buf ^ 1], A + (size_t)m0 * K + kn, g1t);
        tdm_load(Bs[buf ^ 1], Bt + (size_t)n0 * K + kn, g1t);
        __builtin_amdgcn_s_wait_tensorcnt((short)2);
      } else {
        __builtin_amdgcn_s_wait_tensorcnt((short)0);
      }
    }
    __syncthreads();  // current tile visible to all waves
#else
    __syncthreads();
    const int k0 = it << 5;
#pragma unroll
    for (int c = 0; c < 2; ++c) {
      const int idx = t + c * 256;
      const int row = idx >> 2, ch = idx & 3;
      *(bf16x8*)&As[buf][row * LDT + ch * 8] =
          *(const bf16x8*)&A[(size_t)(m0 + row) * K + k0 + ch * 8];
      *(bf16x8*)&Bs[buf][row * LDT + ch * 8] =
          *(const bf16x8*)&Bt[(size_t)(n0 + row) * K + k0 + ch * 8];
    }
    __syncthreads();
#endif

    bf16x16 af[2];
#pragma unroll
    for (int i = 0; i < 2; ++i)
      af[i] = load_frag_a(&As[buf][(wm * 32 + i * 16) * LDT], LDT);
#pragma unroll
    for (int j = 0; j < 4; ++j) {
      bf16x16 bf = load_frag_b_t(&Bs[buf][(wn * 64 + j * 16) * LDT], LDT);
#pragma unroll
      for (int i = 0; i < 2; ++i) acc[i][j] = wmma_bf16(af[i], bf, acc[i][j]);
    }
    __syncthreads();  // reads done before this buffer is overwritten
  }

  // Epilogue. C-layout: VGPR r -> M = r + 8*(lane>=16), N = lane&15.
#pragma unroll
  for (int i = 0; i < 2; ++i)
#pragma unroll
    for (int j = 0; j < 4; ++j) {
      const int rb  = m0 + wm * 32 + i * 16 + 8 * lg;  // rows rb..rb+7
      const int col = n0 + wn * 64 + j * 16 + l15;
      if constexpr (MODE == 1) {
        const int bb = rb >> 10, sb = rb & 1023;  // rb..rb+7 same 1024-block
        const int hh = col >> 6, hd = col & 63;
        const size_t bh = (size_t)(bb * 16 + hh);
        bf16x8 pack;
#pragma unroll
        for (int r = 0; r < 8; ++r) {
          const float v = acc[i][j][r] + bias[col];
          pack[r] = (bf16_t)v;
          ((bf16_t*)outp)[(bh * 1024 + (sb + r)) * 64 + hd] = pack[r];
        }
        // transposed copy: contiguous along sequence -> one 16B store
        *(bf16x8*)&((bf16_t*)outp2)[(bh * 64 + hd) * 1024 + sb] = pack;
      } else {
#pragma unroll
        for (int r = 0; r < 8; ++r) {
          const int row = rb + r;
          const float v = acc[i][j][r] + bias[col];
          if constexpr (MODE == 2) {
            ((float*)outp)[(size_t)row * N + col] =
                v + res[(size_t)row * N + col];
          } else {  // MODE == 3: exact GELU
            const float gl =
                0.5f * v * (1.0f + erff(v * 0.70710678118654752f));
            ((bf16_t*)outp)[(size_t)row * N + col] = (bf16_t)gl;
          }
        }
      }
    }
}

// ---------------------------------------------------------------------------
// Pseudo-MHSA, flash-style, one block per (b, h, 128-query tile).
//   qw = x_q @ Wh ; scores = qw @ x^T / 64 ; online softmax ; O = P @ x
// x in both layouts: xh [B,H,S,64] (scores B-frags contiguous) and
// xhT [B,H,64,S] (PV B-frags contiguous). whT is [H,64(e),64(d)].
// Output block written with one TENSOR_STORE_FROM_LDS.
// ---------------------------------------------------------------------------
__global__ __launch_bounds__(256) void attn_kernel(
    const bf16_t* __restrict__ xh, const bf16_t* __restrict__ xhT,
    const bf16_t* __restrict__ whT, bf16_t* __restrict__ out) {
  constexpr int LDK  = 72;   // 64 + 8 pad (144B rows)
  constexpr int LDKT = 136;  // 128 + 8 pad (272B rows)
  constexpr int LDQ  = 72;
  constexpr int LDP  = 40;
  __shared__ __align__(16) bf16_t xk[128 * LDK];    // K tile [key][hd]
  __shared__ __align__(16) bf16_t xkT[64 * LDKT];   // K tile [hd][key]
  __shared__ __align__(16) bf16_t qw[128 * LDQ];    // Q@Wh; reused for O out
  __shared__ __align__(16) bf16_t ps[8 * 16 * LDP]; // per-wave P staging

  const int bid = blockIdx.x;
  const int qt = bid & 7, h = (bid >> 3) & 15, b = bid >> 7;
  const bf16_t* xbh  = xh  + ((size_t)(b * 16 + h)) * (1024 * 64);
  const bf16_t* xbhT = xhT + ((size_t)(b * 16 + h)) * (64 * 1024);
  const bf16_t* whp  = whT + (size_t)h * (64 * 64);
  const int t = threadIdx.x, w = t >> 5, l = t & 31;
  const int l15 = l & 15, lg = l >> 4;
  const int s0 = qt * 128;
  const f32x8 zero = {0.f, 0.f, 0.f, 0.f, 0.f, 0.f, 0.f, 0.f};

#if USE_TDM
  // xk : 128 rows x 64 el, stride 64   (pad 32 DW -> +4 DW => LDK=72)
  const i32x8 g1k  = tdm_make_g1(64, 128, 64, 1, 4, 3);
  // xkT: 64 rows x 128 el, stride 1024 (pad 64 DW -> +4 DW => LDKT=136)
  const i32x8 g1kT = tdm_make_g1(128, 64, 1024, 1, 5, 3);
#endif

  // --- qw = x_q @ Wh : A/B frags straight from global (contiguous per lane).
  {
    f32x8 qa[4];
#pragma unroll
    for (int j = 0; j < 4; ++j) qa[j] = zero;
#pragma unroll
    for (int kc = 0; kc < 2; ++kc) {
      bf16x16 a = load_frag_a(xbh + (size_t)(s0 + w * 16) * 64 + kc * 32, 64);
#pragma unroll
      for (int j = 0; j < 4; ++j) {
        bf16x16 bb = load_frag_b_t(whp + (j * 16) * 64 + kc * 32, 64);
        qa[j] = wmma_bf16(a, bb, qa[j]);
      }
    }
#pragma unroll
    for (int j = 0; j < 4; ++j)
#pragma unroll
      for (int r = 0; r < 8; ++r)
        qw[(w * 16 + r + 8 * lg) * LDQ + j * 16 + l15] = (bf16_t)qa[j][r];
  }

  float m8[8], s8[8];
#pragma unroll
  for (int r = 0; r < 8; ++r) { m8[r] = -3.0e38f; s8[r] = 0.f; }
  f32x8 Oacc[4];
#pragma unroll
  for (int d = 0; d < 4; ++d) Oacc[d] = zero;

  for (int kt = 0; kt < 8; ++kt) {
    __syncthreads();  // previous tile fully consumed; qw visible on kt==0
#if USE_TDM
    if (w == 0) {
      tdm_load(xk, xbh + (size_t)(kt * 128) * 64, g1k);
      tdm_load(xkT, xbhT + (size_t)(kt * 128), g1kT);
      __builtin_amdgcn_s_wait_tensorcnt((short)0);
    }
#else
#pragma unroll
    for (int c = 0; c < 4; ++c) {
      const int idx = t + c * 256;
      const int row = idx >> 3, ch = idx & 7;
      *(bf16x8*)&xk[row * LDK + ch * 8] =
          *(const bf16x8*)&xbh[(size_t)(kt * 128 + row) * 64 + ch * 8];
      const int rT = idx >> 4, cT = idx & 15;
      *(bf16x8*)&xkT[rT * LDKT + cT * 8] =
          *(const bf16x8*)&xbhT[(size_t)rT * 1024 + kt * 128 + cT * 8];
    }
#endif
    __syncthreads();

    // scores: wave w rows 16w..16w+15 x keys 0..127 (8 N-tiles)
    f32x8 sc[8];
#pragma unroll
    for (int tt = 0; tt < 8; ++tt) sc[tt] = zero;
#pragma unroll
    for (int kc = 0; kc < 2; ++kc) {
      bf16x16 a = load_frag_a(&qw[(w * 16) * LDQ + kc * 32], LDQ);
#pragma unroll
      for (int tt = 0; tt < 8; ++tt) {
        bf16x16 bb = load_frag_b_t(&xk[(tt * 16) * LDK + kc * 32], LDK);
        sc[tt] = wmma_bf16(a, bb, sc[tt]);
      }
    }

    // 1/HD scale + online softmax (row reductions over 16-lane groups)
#pragma unroll
    for (int r = 0; r < 8; ++r) {
      float mx = -3.0e38f;
#pragma unroll
      for (int tt = 0; tt < 8; ++tt) {
        sc[tt][r] *= (1.0f / 64.0f);
        mx = fmaxf(mx, sc[tt][r]);
      }
#pragma unroll
      for (int mask = 1; mask < 16; mask <<= 1)
        mx = fmaxf(mx, __shfl_xor(mx, mask, 32));
      const float mn = fmaxf(m8[r], mx);
      const float al = __expf(m8[r] - mn);
      m8[r] = mn;
      float rsum = 0.f;
#pragma unroll
      for (int tt = 0; tt < 8; ++tt) {
        const float pv = __expf(sc[tt][r] - mn);
        sc[tt][r] = pv;
        rsum += pv;
      }
#pragma unroll
      for (int mask = 1; mask < 16; mask <<= 1)
        rsum += __shfl_xor(rsum, mask, 32);
      s8[r] = s8[r] * al + rsum;
#pragma unroll
      for (int d = 0; d < 4; ++d) Oacc[d][r] *= al;
    }

    // O += P @ V in 4 chunks of 32 keys; P restaged through wave-private
    // LDS (DS ops are in-order within a wave -> no barrier needed).
    bf16_t* pb = ps + w * (16 * LDP);
#pragma unroll
    for (int kc2 = 0; kc2 < 4; ++kc2) {
#pragma unroll
      for (int ttl = 0; ttl < 2; ++ttl) {
        const int tt = kc2 * 2 + ttl;
#pragma unroll
        for (int r = 0; r < 8; ++r)
          pb[(r + 8 * lg) * LDP + ttl * 16 + l15] = (bf16_t)sc[tt][r];
      }
      bf16x16 a = load_frag_a(pb, LDP);
#pragma unroll
      for (int d = 0; d < 4; ++d) {
        bf16x16 bb = load_frag_b_t(&xkT[(d * 16) * LDKT + kc2 * 32], LDKT);
        Oacc[d] = wmma_bf16(a, bb, Oacc[d]);
      }
    }
  }

  // Normalize, restage O into qw (dead now) as a contiguous [128][64] tile,
  // then a single TENSOR_STORE_FROM_LDS writes the block's output
  // (head transpose folded into the 2D descriptor: row stride 1024).
  bf16_t* ob = qw;  // reuse
#pragma unroll
  for (int r = 0; r < 8; ++r) {
    const float inv = 1.0f / s8[r];
    const int ql = w * 16 + r + 8 * lg;  // local query row 0..127
#pragma unroll
    for (int d = 0; d < 4; ++d)
      ob[ql * 64 + d * 16 + l15] = (bf16_t)(Oacc[d][r] * inv);
  }
  __syncthreads();  // all O rows staged
#if USE_TDM
  if (w == 0) {
    const i32x8 g1o = tdm_make_g1(64, 128, 1024, 0, 0, 0);  // no padding
    tdm_store(ob, out + ((size_t)(b * 1024 + s0)) * 1024 + h * 64, g1o);
    __builtin_amdgcn_s_wait_tensorcnt((short)0);  // keep LDS alive until done
  }
#else
#pragma unroll
  for (int c = 0; c < 4; ++c) {
    const int idx = t + c * 256;
    const int row = idx >> 3, ch = idx & 7;
    *(bf16x8*)&out[((size_t)(b * 1024 + s0 + row)) * 1024 + h * 64 + ch * 8] =
        *(const bf16x8*)&ob[row * 64 + ch * 8];
  }
#endif
}

// ---------------------------------------------------------------------------
// Host launch
// ---------------------------------------------------------------------------
extern "C" void kernel_launch(void* const* d_in, const int* in_sizes, int n_in,
                              void* d_out, int out_size, void* d_ws,
                              size_t ws_size, hipStream_t stream) {
  (void)in_sizes; (void)n_in; (void)out_size; (void)ws_size;
  const float* inputs = (const float*)d_in[0];
  const float* W_in   = (const float*)d_in[1];
  const float* b_in   = (const float*)d_in[2];
  const float* attn_w = (const float*)d_in[3];
  const float* W_out  = (const float*)d_in[4];
  const float* b_out  = (const float*)d_in[5];
  const float* W1     = (const float*)d_in[6];
  const float* b1     = (const float*)d_in[7];
  const float* W2     = (const float*)d_in[8];
  const float* b2     = (const float*)d_in[9];
  const float* ln1_g  = (const float*)d_in[10];
  const float* ln1_b  = (const float*)d_in[11];
  const float* ln2_g  = (const float*)d_in[12];
  const float* ln2_b  = (const float*)d_in[13];

  const int T = 8192;  // B*S tokens
  const int D = 1024, MLPD = 4096;

  char* ws = (char*)d_ws;
  size_t off = 0;
  auto alloc = [&](size_t bytes) -> void* {
    void* p = ws + off;
    off += (bytes + 255) & ~(size_t)255;
    return p;
  };
  // Pre-transposed bf16 weights: Wt[N][K]
  bf16_t* wtb_in  = (bf16_t*)alloc((size_t)D * D * 2);
  bf16_t* wtb_out = (bf16_t*)alloc((size_t)D * D * 2);
  bf16_t* wtb_1   = (bf16_t*)alloc((size_t)D * MLPD * 2);
  bf16_t* wtb_2   = (bf16_t*)alloc((size_t)MLPD * D * 2);
  bf16_t* wtb_at  = (bf16_t*)alloc((size_t)16 * 64 * 64 * 2);  // whT[h][e][d]
  bf16_t* xln     = (bf16_t*)alloc((size_t)T * D * 2);
  bf16_t* xhd     = (bf16_t*)alloc((size_t)T * D * 2);  // [B,H,S,hd]
  bf16_t* xhdT    = (bf16_t*)alloc((size_t)T * D * 2);  // [B,H,hd,S]
  bf16_t* attn_o  = (bf16_t*)alloc((size_t)T * D * 2);  // [B,S,D]
  float*  x2      = (float*)alloc((size_t)T * D * 4);   // residual 1
  bf16_t* yln     = (bf16_t*)alloc((size_t)T * D * 2);
  bf16_t* hbuf    = (bf16_t*)alloc((size_t)T * MLPD * 2);

  // Weight transpose+convert (dims are compile-time -> shift addressing)
  cvt_t_kernel<1024, 1024>
      <<<dim3((1024 * 1024 + 255) / 256, 1), 256, 0, stream>>>(W_in, wtb_in);
  cvt_t_kernel<1024, 1024>
      <<<dim3((1024 * 1024 + 255) / 256, 1), 256, 0, stream>>>(W_out, wtb_out);
  cvt_t_kernel<1024, 4096>
      <<<dim3((1024 * 4096 + 255) / 256, 1), 256, 0, stream>>>(W1, wtb_1);
  cvt_t_kernel<4096, 1024>
      <<<dim3((4096 * 1024 + 255) / 256, 1), 256, 0, stream>>>(W2, wtb_2);
  cvt_t_kernel<64, 64>
      <<<dim3((64 * 64 + 255) / 256, 16), 256, 0, stream>>>(attn_w, wtb_at);

  // LN1
  ln_kernel<<<T, 256, 0, stream>>>(inputs, ln1_g, ln1_b, xln);
  // x = LN1(x) @ W_in + b_in -> head layout (both orientations)
  gemm_kernel<1><<<dim3(D / 128, T / 128), 256, 0, stream>>>(
      xln, wtb_in, b_in, nullptr, xhd, xhdT, T, D, D);
  // attention
  attn_kernel<<<8 * 16 * 8, 256, 0, stream>>>(xhd, xhdT, wtb_at, attn_o);
  // x2 = attn @ W_out + b_out + inputs (fp32 residual)
  gemm_kernel<2><<<dim3(D / 128, T / 128), 256, 0, stream>>>(
      attn_o, wtb_out, b_out, inputs, x2, nullptr, T, D, D);
  // LN2
  ln_kernel<<<T, 256, 0, stream>>>(x2, ln2_g, ln2_b, yln);
  // h = gelu(y @ W1 + b1)
  gemm_kernel<3><<<dim3(MLPD / 128, T / 128), 256, 0, stream>>>(
      yln, wtb_1, b1, nullptr, hbuf, nullptr, T, MLPD, D);
  // out = h @ W2 + b2 + x2
  gemm_kernel<2><<<dim3(D / 128, T / 128), 256, 0, stream>>>(
      hbuf, wtb_2, b2, x2, (float*)d_out, nullptr, T, D, MLPD);
}
